// GCN_dgl_32255204393049
// MI455X (gfx1250) — matrix operand order repaired and checked by reading, not compiled, not run on hardware
//
#include <hip/hip_runtime.h>

typedef __attribute__((ext_vector_type(16))) _Float16 v16h;
typedef __attribute__((ext_vector_type(8)))  float    v8f;

#define BT 256

// ---------- small setup kernels ----------
__global__ void k_zero2(float* __restrict__ a, float* __restrict__ b, int n) {
    int i = blockIdx.x * blockDim.x + threadIdx.x;
    if (i < n) { a[i] = 0.f; b[i] = 0.f; }
}

__global__ void k_degrees(const int* __restrict__ src, const int* __restrict__ dst,
                          float* __restrict__ degout, float* __restrict__ degin, int E) {
    int e = blockIdx.x * blockDim.x + threadIdx.x;
    if (e < E) {
        atomicAdd(&degout[src[e]], 1.0f);
        atomicAdd(&degin[dst[e]], 1.0f);
    }
}

__global__ void k_norm(float* __restrict__ ns, float* __restrict__ nd, int n) {
    int i = blockIdx.x * blockDim.x + threadIdx.x;
    if (i < n) {
        ns[i] = rsqrtf(fmaxf(ns[i], 1.0f));
        nd[i] = rsqrtf(fmaxf(nd[i], 1.0f));
    }
}

// Pack weight matrix W[128 x Fout] (f32) into WMMA B-fragment-major f16:
// Bp[nt][chunk][lane][e2], where lane holds column nt*16+(lane&15),
// K = chunk*32 + (e2&7) + (lane>>4)*8 + ((e2&8)<<1)  (matches A interleave).
__global__ void k_packB(const float* __restrict__ W, _Float16* __restrict__ Bp,
                        int Fout, int nfrag) {
    int i = blockIdx.x * blockDim.x + threadIdx.x;   // fragment-lane id
    if (i >= nfrag) return;
    int lane = i & 31, chunk = (i >> 5) & 3, nt = i >> 7;
    int hl = lane & 15, hi = lane >> 4;
    int col = (nt << 4) + hl;
    int k0  = (chunk << 5) + (hi << 3);
    v16h frag;
#pragma unroll
    for (int e2 = 0; e2 < 16; ++e2)
        frag[e2] = (_Float16)W[(size_t)(k0 + (e2 & 7) + ((e2 & 8) << 1)) * Fout + col];
    ((v16h*)Bp)[i] = frag;
}

// ---------- per-layer kernels ----------
// xs[n,f] = h[n,f] * norm_src[n]; also zero agg (same shape) for the scatter.
__global__ void k_prep(const float* __restrict__ hin, const float* __restrict__ ns,
                       float* __restrict__ xs, float* __restrict__ agg, int total) {
    int i = blockIdx.x * blockDim.x + threadIdx.x;
    if (i < total) {
        int n = i >> 7;           // F = 128
        xs[i]  = hin[i] * ns[n];
        agg[i] = 0.f;
    }
}

// One wave32 per edge; edge id is wave-uniform -> src/dst/w become s_load.
// Each lane handles 4 consecutive features (32*4 = 128).
__global__ void k_scatter(const float* __restrict__ xs, const float* __restrict__ w,
                          const int* __restrict__ src, const int* __restrict__ dst,
                          float* __restrict__ agg, int E) {
    int e = blockIdx.x * (blockDim.x >> 5) + (threadIdx.x >> 5);  // uniform per wave
    if (e >= E) return;
    int lane = threadIdx.x & 31;
    int s = src[e], d = dst[e];
    float we = w[e];
    float4 v = ((const float4*)(xs + (long)s * 128))[lane];
    float* drow = agg + (long)d * 128 + lane * 4;
    atomicAdd(drow + 0, v.x * we);
    atomicAdd(drow + 1, v.y * we);
    atomicAdd(drow + 2, v.z * we);
    atomicAdd(drow + 3, v.w * we);
}

// Scale by norm_dst, convert to f16 and pack directly into WMMA A-fragment-major:
// Ap[mt][chunk][lane][e2]; row n = mt*16+(lane&15);
// K = chunk*32 + (e2&7) + (lane>>4)*8 + ((e2&8)<<1)   (ISA 7.12.2 interleave).
// One thread per fragment-lane: reads two contiguous 32B spans of the agg row.
__global__ void k_post_pack(const float* __restrict__ agg, const float* __restrict__ nd,
                            _Float16* __restrict__ Ap, int nfrag) {
    int i = blockIdx.x * blockDim.x + threadIdx.x;   // fragment-lane id
    if (i >= nfrag) return;
    int lane = i & 31, chunk = (i >> 5) & 3, mt = i >> 7;
    int hl = lane & 15, hi = lane >> 4;
    int n = (mt << 4) + hl;
    float s = nd[n];
    const float* arow = agg + (size_t)n * 128 + (chunk << 5);
    v16h frag;
#pragma unroll
    for (int e2 = 0; e2 < 16; ++e2) {
        int kin = (e2 & 7) + (hi << 3) + ((e2 & 8) << 1);
        frag[e2] = (_Float16)(arow[kin] * s);
    }
    ((v16h*)Ap)[i] = frag;
}

// WMMA GEMM on pre-packed fragments. One wave computes a 16-row strip across
// all Fout = NT*16 columns: A fragment loaded once per k-chunk, reused NT times.
template <int NT, bool RELU>
__global__ void k_gemm_wmma(const _Float16* __restrict__ Ap, const _Float16* __restrict__ Bp,
                            const float* __restrict__ bias, float* __restrict__ out,
                            int ntiles_m) {
    constexpr int Fout = NT * 16;
    int lane = threadIdx.x & 31;
    int wave = threadIdx.x >> 5;
    int mt = blockIdx.x * (blockDim.x >> 5) + wave;   // wave-uniform -> EXEC all-1s
    if (mt >= ntiles_m) return;

    v8f zero = {};
    v8f acc[NT];
#pragma unroll
    for (int t = 0; t < NT; ++t) acc[t] = zero;

    const v16h* Afrag = (const v16h*)Ap + (size_t)mt * 4 * 32 + lane;  // [mt][c][lane]
    const v16h* Bfrag = (const v16h*)Bp + lane;                        // [nt][c][lane]
#pragma unroll
    for (int c = 0; c < 4; ++c) {
        v16h a = Afrag[c * 32];
#pragma unroll
        for (int t = 0; t < NT; ++t) {
            v16h b = Bfrag[(size_t)(t * 4 + c) * 32];
            acc[t] = __builtin_amdgcn_wmma_f32_16x16x32_f16(false, a, false, b,
                                                            (short)0, acc[t], false, false);
        }
    }

    int hl = lane & 15, hi = lane >> 4;
#pragma unroll
    for (int t = 0; t < NT; ++t) {
        int col = t * 16 + hl;
        float bv = bias[col];
#pragma unroll
        for (int v = 0; v < 8; ++v) {
            int row = (mt << 4) + v + hi * 8;             // C/D: VGPR v -> M=v(+8)
            float val = acc[t][v] + bv;
            if (RELU) val = fmaxf(val, 0.f);
            out[(size_t)row * Fout + col] = val;
        }
    }
}

// One wave32 per row of 64 logits (2 per lane); shuffle-reduce max and sum.
__global__ void k_logsoftmax(const float* __restrict__ logits, float* __restrict__ out, int N) {
    int lane = threadIdx.x & 31;
    int wave = threadIdx.x >> 5;
    int row  = blockIdx.x * (blockDim.x >> 5) + wave;
    if (row >= N) return;
    const float* r = logits + (long)row * 64;
    float v0 = r[lane], v1 = r[lane + 32];
    float m = fmaxf(v0, v1);
#pragma unroll
    for (int off = 16; off > 0; off >>= 1) m = fmaxf(m, __shfl_xor(m, off, 32));
    float s = expf(v0 - m) + expf(v1 - m);
#pragma unroll
    for (int off = 16; off > 0; off >>= 1) s += __shfl_xor(s, off, 32);
    float lse = m + logf(s);
    out[(long)row * 64 + lane]      = v0 - lse;
    out[(long)row * 64 + lane + 32] = v1 - lse;
}

extern "C" void kernel_launch(void* const* d_in, const int* in_sizes, int n_in,
                              void* d_out, int out_size, void* d_ws, size_t ws_size,
                              hipStream_t stream) {
    const float* x  = (const float*)d_in[0];
    const float* w  = (const float*)d_in[1];
    const float* W0 = (const float*)d_in[2];
    const float* b0 = (const float*)d_in[3];
    const float* W1 = (const float*)d_in[4];
    const float* b1 = (const float*)d_in[5];
    const float* W2 = (const float*)d_in[6];
    const float* b2 = (const float*)d_in[7];
    const int* src  = (const int*)d_in[8];
    const int* dst  = (const int*)d_in[9];

    const int N    = in_sizes[0] / 128;   // 50000 (divisible by 16)
    const int E    = in_sizes[1];         // 800000
    const int OUTC = in_sizes[7];         // 64

    // workspace carve-up
    float* ws       = (float*)d_ws;
    float* norm_src = ws;
    float* norm_dst = norm_src + N;
    float* xs       = norm_dst + N;
    float* agg      = xs   + (size_t)N * 128;
    float* hbuf     = agg  + (size_t)N * 128;
    _Float16* Ap  = (_Float16*)(hbuf + (size_t)N * 128);   // N*128 halves, 32B aligned
    _Float16* W0p = Ap  + (size_t)N * 128;                  // 8*4*32*16 = 16384 halves
    _Float16* W1p = W0p + 128 * 128;
    _Float16* W2p = W1p + 128 * 128;

    const int gN  = (N + BT - 1) / BT;
    const int gE  = (E + BT - 1) / BT;
    const int totalNF = N * 128;
    const int gNF = (totalNF + BT - 1) / BT;
    const int gEdge = (E + 7) / 8;          // 8 waves (edges) per 256-thread block
    const int mt = N / 16;
    const int nfragA = mt * 4 * 32;         // fragment-lanes in packed A
    const int gFA = (nfragA + BT - 1) / BT;

    // degrees -> norms
    k_zero2  <<<gN, BT, 0, stream>>>(norm_src, norm_dst, N);
    k_degrees<<<gE, BT, 0, stream>>>(src, dst, norm_src, norm_dst, E);
    k_norm   <<<gN, BT, 0, stream>>>(norm_src, norm_dst, N);

    // weights -> packed f16 B fragments
    k_packB<<<(8 * 4 * 32 + BT - 1) / BT, BT, 0, stream>>>(W0, W0p, 128, 8 * 4 * 32);
    k_packB<<<(8 * 4 * 32 + BT - 1) / BT, BT, 0, stream>>>(W1, W1p, 128, 8 * 4 * 32);
    k_packB<<<(4 * 4 * 32 + BT - 1) / BT, BT, 0, stream>>>(W2, W2p, OUTC, 4 * 4 * 32);

    // layer 0
    k_prep     <<<gNF,   BT, 0, stream>>>(x, norm_src, xs, agg, totalNF);
    k_scatter  <<<gEdge, BT, 0, stream>>>(xs, w, src, dst, agg, E);
    k_post_pack<<<gFA,   BT, 0, stream>>>(agg, norm_dst, Ap, nfragA);
    k_gemm_wmma<8, true><<<(mt + 7) / 8, 256, 0, stream>>>(Ap, W0p, b0, hbuf, mt);

    // layer 1
    k_prep     <<<gNF,   BT, 0, stream>>>(hbuf, norm_src, xs, agg, totalNF);
    k_scatter  <<<gEdge, BT, 0, stream>>>(xs, w, src, dst, agg, E);
    k_post_pack<<<gFA,   BT, 0, stream>>>(agg, norm_dst, Ap, nfragA);
    k_gemm_wmma<8, true><<<(mt + 7) / 8, 256, 0, stream>>>(Ap, W1p, b1, hbuf, mt);

    // layer 2 (logits into xs, which the scatter no longer needs)
    k_prep     <<<gNF,   BT, 0, stream>>>(hbuf, norm_src, xs, agg, totalNF);
    k_scatter  <<<gEdge, BT, 0, stream>>>(xs, w, src, dst, agg, E);
    k_post_pack<<<gFA,   BT, 0, stream>>>(agg, norm_dst, Ap, nfragA);
    k_gemm_wmma<4, false><<<(mt + 7) / 8, 256, 0, stream>>>(Ap, W2p, b2, xs, mt);

    k_logsoftmax<<<(N + 7) / 8, 256, 0, stream>>>(xs, (float*)d_out, N);
}
